// MultiHeadAttention_69303592288919
// MI455X (gfx1250) — matrix operand order repaired and checked by reading, not compiled, not run on hardware
//
#include <hip/hip_runtime.h>

// ---------------------------------------------------------------------------
// MHA for N=2, S=2048, E=1024, H=16, HD=64 on gfx1250 (MI455X).
// fp32 -> bf16 once; all matmuls via v_wmma_f32_16x16x32_bf16.
// GEMMs: 128x64 block tile, wave tile 32x64 (2 A-frags x 4 B-frags = 8 WMMA
// per K-step); B tile double-buffered in LDS via global_load_async_to_lds_b128
// (ASYNCcnt) with register-prefetched A fragments.
// Attention: causal flash-attention with WMMA for QK^T and PV.
// ---------------------------------------------------------------------------

#define N_BATCH 2
#define SEQ     2048
#define EMB     1024
#define NHEAD   16
#define HDIM    64
#define FQKV    (3 * EMB)        // 3072
#define MTOT    (N_BATCH * SEQ)  // 4096

typedef __attribute__((ext_vector_type(16))) __bf16 bf16x16;
typedef __attribute__((ext_vector_type(8)))  __bf16 bf16x8;
typedef __attribute__((ext_vector_type(8)))  float  f32x8;

__device__ __forceinline__ f32x8 wmma_bf16f32(bf16x16 a, bf16x16 b, f32x8 c) {
  // (neg_a, A, neg_b, B, c_mod, C, reuse_a, reuse_b)
  return __builtin_amdgcn_wmma_f32_16x16x32_bf16(false, a, false, b, (short)0, c,
                                                 false, false);
}

// A-fragment (16x32 bf16, M x K). Lanes 0-15 -> M=lane, K 0..15;
// lanes 16-31 -> M=lane-16, K 16..31. Two 16-byte loads per lane.
__device__ __forceinline__ bf16x16 load_a_frag(const __bf16* base, int ldk,
                                               int k0, int lane) {
  const int row  = lane & 15;
  const int koff = (lane < 16) ? 0 : 16;
  const __bf16* p = base + (size_t)row * ldk + k0 + koff;
  bf16x8 lo = *reinterpret_cast<const bf16x8*>(p);
  bf16x8 hi = *reinterpret_cast<const bf16x8*>(p + 8);
  bf16x16 a;
#pragma unroll
  for (int i = 0; i < 8; ++i) { a[i] = lo[i]; a[i + 8] = hi[i]; }
  return a;
}

// B-fragment (32x16 bf16, K x N): memory holds column n's K contiguously
// (B^T row-major, stride ldk). Lanes 0-15: K=0..15 of col=lane; lanes
// 16-31: K=16..31 of col=lane-16. One 32-byte load per lane.
__device__ __forceinline__ bf16x16 load_b_frag(const __bf16* base, int ldk,
                                               int k0, int lane) {
  const int col = lane & 15;
  const int kb  = (lane < 16) ? 0 : 16;
  const __bf16* p = base + (size_t)col * ldk + k0 + kb;
  return *reinterpret_cast<const bf16x16*>(p);
}

// ---------------------------------------------------------------------------
// Async global->LDS staging (gfx1250 async path, ASYNCcnt-tracked).
// Stages a 64(col) x 32(K) bf16 tile (4 KB) = 256 x b128; 128 threads x 2.
// LDS layout: lds[col*32 + k], identical sub-block of the global layout.
// ---------------------------------------------------------------------------
__device__ __forceinline__ void stage_b_tile_async(__bf16* lds_buf,
                                                   const __bf16* gbase,
                                                   int ldk, int tid) {
#pragma unroll
  for (int r = 0; r < 2; ++r) {
    const int u  = tid + r * 128;   // 0..255
    const int c  = u >> 2;          // col 0..63
    const int k8 = (u & 3) * 8;     // k 0,8,16,24
    // addrspace(3) allocations live at addr[31:0] of the generic pointer;
    // the async op's VDST takes the 32-bit LDS byte address.
    const unsigned lds_addr = (unsigned)(size_t)(lds_buf + c * 32 + k8);
    const __bf16* g = gbase + (size_t)c * ldk + k8;
    asm volatile("global_load_async_to_lds_b128 %0, %1, off"
                 :: "v"(lds_addr), "v"(g)
                 : "memory");
  }
}

__device__ __forceinline__ void wait_async_le2() {
  asm volatile("s_wait_asynccnt 0x2" ::: "memory");
}
__device__ __forceinline__ void wait_async_0() {
  asm volatile("s_wait_asynccnt 0x0" ::: "memory");
}

// ---------------------------------------------------------------------------
// Shared GEMM core. Block = 128 thr (4 waves): block tile 128(M) x 64(F);
// wave w owns rows m0..m0+31 (2 A-frags), all 64 cols (4 B-frags) -> 8 WMMA
// per K-step, each LDS B-frag feeding two WMMAs. B tile double-buffered in
// LDS via async loads; A fragments register-prefetched one K-step ahead.
// ---------------------------------------------------------------------------
__device__ __forceinline__ void gemm_core(const __bf16* abase, int lda,
                                          const __bf16* wbase_f0, int ldw,
                                          int kdim, __bf16 (*Bl)[64 * 32],
                                          int tid, int lane,
                                          f32x8 acc[2][4]) {
  const int ksteps = kdim / 32;
  stage_b_tile_async(&Bl[0][0], wbase_f0, ldw, tid);
  bf16x16 a_cur0 = load_a_frag(abase, lda, 0, lane);
  bf16x16 a_cur1 = load_a_frag(abase + (size_t)16 * lda, lda, 0, lane);

#pragma unroll 2
  for (int i = 0; i < ksteps; ++i) {
    bf16x16 a_next0 = a_cur0, a_next1 = a_cur1;
    if (i + 1 < ksteps) {
      const int nk = (i + 1) * 32;
      stage_b_tile_async(&Bl[(i + 1) & 1][0], wbase_f0 + nk, ldw, tid);
      a_next0 = load_a_frag(abase, lda, nk, lane);
      a_next1 = load_a_frag(abase + (size_t)16 * lda, lda, nk, lane);
      wait_async_le2();   // stage i complete; stage i+1 (2 ops) in flight
    } else {
      wait_async_0();
    }
    __syncthreads();      // stage-i tile visible to all waves
    const __bf16* bbuf = &Bl[i & 1][0];
#pragma unroll
    for (int t = 0; t < 4; ++t) {
      bf16x16 b = load_b_frag(bbuf + t * 16 * 32, 32, 0, lane);
      acc[0][t] = wmma_bf16f32(a_cur0, b, acc[0][t]);
      acc[1][t] = wmma_bf16f32(a_cur1, b, acc[1][t]);
    }
    __syncthreads();      // done reading buf before it is restaged
    a_cur0 = a_next0;
    a_cur1 = a_next1;
  }
}

// ---------------------------------------------------------------------------
// fp32 -> bf16 conversion
// ---------------------------------------------------------------------------
__global__ void cvt_f32_bf16_kernel(const float* __restrict__ in,
                                    __bf16* __restrict__ out, int n) {
  int i = blockIdx.x * blockDim.x + threadIdx.x;
  if (i < n) out[i] = (__bf16)in[i];
}

// ---------------------------------------------------------------------------
// QKV projection: qkv[m, f] = sum_k x[m,k] * w[f,k]; scatters bf16 Q/K/V
// into [n, h, s, d] layout.
// ---------------------------------------------------------------------------
__global__ void __launch_bounds__(128)
qkv_gemm_kernel(const __bf16* __restrict__ xb, const __bf16* __restrict__ wb,
                __bf16* __restrict__ qb, __bf16* __restrict__ kb,
                __bf16* __restrict__ vb) {
  __shared__ __align__(64) __bf16 Bl[2][64 * 32];
  const int tid  = threadIdx.x;
  const int lane = tid & 31;
  const int wave = tid >> 5;
  const int m0 = blockIdx.y * 128 + wave * 32;
  const int f0 = blockIdx.x * 64;

  f32x8 acc[2][4];
#pragma unroll
  for (int r = 0; r < 2; ++r)
#pragma unroll
    for (int t = 0; t < 4; ++t)
#pragma unroll
      for (int v = 0; v < 8; ++v) acc[r][t][v] = 0.0f;

  gemm_core(xb + (size_t)m0 * EMB, EMB, wb + (size_t)f0 * EMB, EMB, EMB, Bl,
            tid, lane, acc);

  const int g   = (lane >= 16) ? 8 : 0;
  const int col = lane & 15;
#pragma unroll
  for (int r = 0; r < 2; ++r) {
#pragma unroll
    for (int t = 0; t < 4; ++t) {
      const int f = f0 + t * 16 + col;
      const int h = f / (3 * HDIM);
      const int j = f % (3 * HDIM);
#pragma unroll
      for (int v = 0; v < 8; ++v) {
        const int m = m0 + r * 16 + v + g;
        const int n = m >> 11;       // / SEQ
        const int s = m & (SEQ - 1); // % SEQ
        const __bf16 val = (__bf16)acc[r][t][v];
        const size_t idx = (((size_t)n * NHEAD + h) * SEQ + s) * HDIM;
        if (j < HDIM)            qb[idx + j] = val;
        else if (j < 2 * HDIM)   kb[idx + j - HDIM] = val;
        else                     vb[idx + j - 2 * HDIM] = val;
      }
    }
  }
}

// ---------------------------------------------------------------------------
// Output projection: out[m, f] = sum_e vals[m,e] * o_w[f,e], fp32 out.
// ---------------------------------------------------------------------------
__global__ void __launch_bounds__(128)
out_gemm_kernel(const __bf16* __restrict__ ab, const __bf16* __restrict__ wb,
                float* __restrict__ out) {
  __shared__ __align__(64) __bf16 Bl[2][64 * 32];
  const int tid  = threadIdx.x;
  const int lane = tid & 31;
  const int wave = tid >> 5;
  const int m0 = blockIdx.y * 128 + wave * 32;
  const int f0 = blockIdx.x * 64;

  f32x8 acc[2][4];
#pragma unroll
  for (int r = 0; r < 2; ++r)
#pragma unroll
    for (int t = 0; t < 4; ++t)
#pragma unroll
      for (int v = 0; v < 8; ++v) acc[r][t][v] = 0.0f;

  gemm_core(ab + (size_t)m0 * EMB, EMB, wb + (size_t)f0 * EMB, EMB, EMB, Bl,
            tid, lane, acc);

  const int g   = (lane >= 16) ? 8 : 0;
  const int col = lane & 15;
#pragma unroll
  for (int r = 0; r < 2; ++r) {
#pragma unroll
    for (int t = 0; t < 4; ++t) {
      const int f = f0 + t * 16 + col;
#pragma unroll
      for (int v = 0; v < 8; ++v) {
        const int m = m0 + r * 16 + v + g;
        out[(size_t)m * EMB + f] = acc[r][t][v];
      }
    }
  }
}

// ---------------------------------------------------------------------------
// Flash attention (causal). Grid: (SEQ/64, N*H). Block = 128 thr (4 waves);
// each wave owns 16 query rows. Key blocks of 32; online softmax; P staged
// through LDS into A-fragment layout; V block staged transposed in LDS so
// P*V B-fragments are contiguous ds loads.
// ---------------------------------------------------------------------------
__global__ void __launch_bounds__(128)
attn_kernel(const __bf16* __restrict__ qb, const __bf16* __restrict__ kb,
            const __bf16* __restrict__ vb, __bf16* __restrict__ valsb) {
  __shared__ __align__(64) __bf16 VT[HDIM][32];   // V^T tile: [d][key]
  __shared__ __align__(64) __bf16 P[4][16][32];   // per-wave P tiles

  const int lane = threadIdx.x & 31;
  const int wave = threadIdx.x >> 5;
  const int nh = blockIdx.y;
  const int n  = nh >> 4;
  const int h  = nh & (NHEAD - 1);
  const int q0 = blockIdx.x * 64;
  const int qw = q0 + wave * 16;            // this wave's first query row
  const int g   = (lane >= 16) ? 8 : 0;
  const int col = lane & 15;
  const float scale = 0.125f;               // 1/sqrt(64)

  const __bf16* qbase = qb + ((size_t)nh * SEQ + qw) * HDIM;
  const bf16x16 qa0 = load_a_frag(qbase, HDIM, 0, lane);
  const bf16x16 qa1 = load_a_frag(qbase, HDIM, 32, lane);

  float m_r[8], l_r[8];
#pragma unroll
  for (int v = 0; v < 8; ++v) { m_r[v] = -__builtin_inff(); l_r[v] = 0.0f; }
  f32x8 o[4];
#pragma unroll
  for (int t = 0; t < 4; ++t)
#pragma unroll
    for (int v = 0; v < 8; ++v) o[t][v] = 0.0f;

  const int jmax = q0 + 63;                 // last key any row in block needs
  for (int j0 = 0; j0 <= jmax; j0 += 32) {
    __syncthreads();
    {   // cooperative transpose-stage of V[j0..j0+31][0..63] into VT[d][key]
      const int key = threadIdx.x >> 2;
      const int dg  = (threadIdx.x & 3) * 16;
      const __bf16* vp = vb + ((size_t)nh * SEQ + (j0 + key)) * HDIM + dg;
      bf16x8 r0 = *reinterpret_cast<const bf16x8*>(vp);
      bf16x8 r1 = *reinterpret_cast<const bf16x8*>(vp + 8);
#pragma unroll
      for (int i = 0; i < 8; ++i) {
        VT[dg + i][key]     = r0[i];
        VT[dg + 8 + i][key] = r1[i];
      }
    }
    __syncthreads();

    if (j0 <= qw + 15) {   // wave-uniform predicate: EXEC stays full inside
      // ---- S = Q K^T for keys [j0, j0+32)
      f32x8 s0, s1;
#pragma unroll
      for (int v = 0; v < 8; ++v) { s0[v] = 0.0f; s1[v] = 0.0f; }
      const __bf16* kbase0 = kb + ((size_t)nh * SEQ + j0) * HDIM;
      const __bf16* kbase1 = kbase0 + 16 * HDIM;
      s0 = wmma_bf16f32(qa0, load_b_frag(kbase0, HDIM, 0, lane), s0);
      s0 = wmma_bf16f32(qa1, load_b_frag(kbase0, HDIM, 32, lane), s0);
      s1 = wmma_bf16f32(qa0, load_b_frag(kbase1, HDIM, 0, lane), s1);
      s1 = wmma_bf16f32(qa1, load_b_frag(kbase1, HDIM, 32, lane), s1);

      // ---- mask + online softmax (per row = (v, lane-group))
#pragma unroll
      for (int v = 0; v < 8; ++v) {
        const int qr = qw + v + g;
        float x0 = (j0 + col <= qr)      ? s0[v] * scale : -__builtin_inff();
        float x1 = (j0 + 16 + col <= qr) ? s1[v] * scale : -__builtin_inff();
        float t = fmaxf(x0, x1);
#pragma unroll
        for (int off = 8; off >= 1; off >>= 1)
          t = fmaxf(t, __shfl_xor(t, off, 32));
        const float mn   = fmaxf(m_r[v], t);
        const float corr = __expf(m_r[v] - mn);
        const float p0   = __expf(x0 - mn);
        const float p1   = __expf(x1 - mn);
        float rs = p0 + p1;
#pragma unroll
        for (int off = 8; off >= 1; off >>= 1)
          rs += __shfl_xor(rs, off, 32);
        l_r[v] = l_r[v] * corr + rs;
        m_r[v] = mn;
#pragma unroll
        for (int t4 = 0; t4 < 4; ++t4) o[t4][v] *= corr;
        const int prow = v + g;
        P[wave][prow][col]      = (__bf16)p0;
        P[wave][prow][col + 16] = (__bf16)p1;
      }

      // ---- O += P V  (A = P 16x32 from LDS, B = V^T tiles from LDS)
      bf16x16 pa = load_a_frag(&P[wave][0][0], 32, 0, lane);
#pragma unroll
      for (int t4 = 0; t4 < 4; ++t4) {
        bf16x16 vbf = load_b_frag(&VT[t4 * 16][0], 32, 0, lane);
        o[t4] = wmma_bf16f32(pa, vbf, o[t4]);
      }
    }
  }

  // ---- epilogue: normalize and write vals[n, s, h*64 + d] in bf16
  float inv_l[8];
#pragma unroll
  for (int v = 0; v < 8; ++v) inv_l[v] = 1.0f / l_r[v];
#pragma unroll
  for (int t4 = 0; t4 < 4; ++t4) {
#pragma unroll
    for (int v = 0; v < 8; ++v) {
      const int qr = qw + v + g;
      const int e  = h * HDIM + t4 * 16 + col;
      valsb[((size_t)n * SEQ + qr) * EMB + e] = (__bf16)(o[t4][v] * inv_l[v]);
    }
  }
}

// ---------------------------------------------------------------------------
// Launch: inputs [x (f32), attn_mask (i32, causal triu -> positional test),
// qkv_w (f32), o_w (f32)]; output f32 [N,S,E].
// ---------------------------------------------------------------------------
extern "C" void kernel_launch(void* const* d_in, const int* in_sizes, int n_in,
                              void* d_out, int out_size, void* d_ws,
                              size_t ws_size, hipStream_t stream) {
  (void)in_sizes; (void)n_in; (void)out_size; (void)ws_size;
  const float* x     = (const float*)d_in[0];
  const float* qkv_w = (const float*)d_in[2];
  const float* o_w   = (const float*)d_in[3];
  float* out = (float*)d_out;

  char* ws = (char*)d_ws;
  size_t off = 0;
  auto carve = [&](size_t bytes) -> void* {
    void* p = ws + off;
    off += (bytes + 255) & ~(size_t)255;
    return p;
  };
  const size_t NSE  = (size_t)MTOT * EMB;
  const size_t NHSD = (size_t)N_BATCH * NHEAD * SEQ * HDIM;
  __bf16* xb    = (__bf16*)carve(NSE * 2);
  __bf16* wqkvb = (__bf16*)carve((size_t)FQKV * EMB * 2);
  __bf16* wob   = (__bf16*)carve((size_t)EMB * EMB * 2);
  __bf16* qbuf  = (__bf16*)carve(NHSD * 2);
  __bf16* kbuf  = (__bf16*)carve(NHSD * 2);
  __bf16* vbuf  = (__bf16*)carve(NHSD * 2);
  __bf16* vals  = (__bf16*)carve(NSE * 2);

  {
    int n0 = (int)NSE;
    cvt_f32_bf16_kernel<<<(n0 + 255) / 256, 256, 0, stream>>>(x, xb, n0);
    int n1 = FQKV * EMB;
    cvt_f32_bf16_kernel<<<(n1 + 255) / 256, 256, 0, stream>>>(qkv_w, wqkvb, n1);
    int n2 = EMB * EMB;
    cvt_f32_bf16_kernel<<<(n2 + 255) / 256, 256, 0, stream>>>(o_w, wob, n2);
  }
  qkv_gemm_kernel<<<dim3(FQKV / 64, MTOT / 128), 128, 0, stream>>>(
      xb, wqkvb, qbuf, kbuf, vbuf);
  attn_kernel<<<dim3(SEQ / 64, N_BATCH * NHEAD), 128, 0, stream>>>(
      qbuf, kbuf, vbuf, vals);
  out_gemm_kernel<<<dim3(EMB / 64, MTOT / 128), 128, 0, stream>>>(vals, wob, out);
}
